// RecurrentNeuralNetwork_47863115547252
// MI455X (gfx1250) — matrix operand order repaired and checked by reading, not compiled, not run on hardware
//
#include <hip/hip_runtime.h>
#include <hip/hip_bf16.h>
#include <math.h>

#define N_HID   256
#define N_IN    2
#define N_OUT   2
#define BATCH   256
#define LENGTH  1024
#define ALPHA   0.25f
#define SIGMA_NEU 0.05f
#define WIN     128            // x prefetch window (timesteps)

typedef __attribute__((ext_vector_type(16))) __bf16 v16bf;
typedef __attribute__((ext_vector_type(8)))  float  v8f;

union Frag16 { v16bf v; uint4 q[2]; __bf16 e[16]; };
union Acc8   { v8f  v; float e[8]; };

__device__ __forceinline__ unsigned hash_u32(unsigned x) {
    x ^= x >> 16; x *= 0x7feb352dU;
    x ^= x >> 15; x *= 0x846ca68bU;
    x ^= x >> 16;
    return x;
}

// CLT Gaussian from one LCG draw: sum of 4 uniform bytes ~ N(510, 147.8^2).
__device__ __forceinline__ float byte_gauss(unsigned v) {
#if __has_builtin(__builtin_amdgcn_sad_u8)
    unsigned s4 = __builtin_amdgcn_sad_u8(v, 0u, 0u);   // v_sad_u8: sum of bytes
#else
    unsigned s4 = (v & 255u) + ((v >> 8) & 255u) + ((v >> 16) & 255u) + (v >> 24);
#endif
    return ((float)(int)s4 - 510.0f) * (1.0f / 147.79f);
}

__device__ __forceinline__ float fast_tanh(float x) {
#if __has_builtin(__builtin_amdgcn_tanhf)
    return __builtin_amdgcn_tanhf(x);                    // v_tanh_f32
#else
    return tanhf(x);
#endif
}

// CDNA5 async global -> LDS copy (ASYNCcnt) via inline asm.
__device__ __forceinline__ void async_load_b128(unsigned lds_off, const float* gptr) {
    asm volatile("global_load_async_to_lds_b128 %0, %1, off"
                 :: "v"(lds_off), "v"((unsigned long long)(uintptr_t)gptr)
                 : "memory");
}
__device__ __forceinline__ void wait_asynccnt0() {
    asm volatile("s_wait_asynccnt 0x0" ::: "memory");
}

// ---------------------------------------------------------------------------
// Main recurrence kernel.
// grid = 16 blocks (batch tiles of 16), block = 512 threads = 16 waves.
// Wave w owns hidden columns [w*16, w*16+16); lane L owns column w*16+(L&15),
// rows m = r + 8*(L>>4) (16x16 f32 C/D WMMA layout) -> hidden state h lives
// permanently in 8 VGPRs per lane; W_hh lives permanently in 64 VGPRs (bf16).
// ---------------------------------------------------------------------------
__global__ __launch_bounds__(512)
void rnn_scan_kernel(const float* __restrict__ x,        // [B, T, N_IN]
                     const float* __restrict__ h0,       // [B, H]
                     const float* __restrict__ w_in,     // [H, N_IN]
                     const float* __restrict__ abs_w0,   // [H, H]
                     const float* __restrict__ w_sign,   // [H, H]
                     const float* __restrict__ is_con,   // [H, H]
                     float* __restrict__ hidden_list)    // [B, T, H]
{
    __shared__ __align__(16) __bf16 a_buf[2][16][N_HID];      // 16 KB, double buffered
    __shared__ __align__(16) float  x_win[2][16][WIN][N_IN];  // 32 KB, double buffered

    const int tid    = threadIdx.x;
    const int wave   = tid >> 5;             // 0..15 : N tile index
    const int lane   = tid & 31;
    const int half   = lane >> 4;            // 0/1
    const int l15    = lane & 15;
    const int batch0 = blockIdx.x * 16;

    const int n = wave * 16 + l15;           // hidden column owned by this lane

    // ---- One-time: resident B fragments (W_hh rows -> bf16 VGPRs) ----
    // B[k,n] = W_hh[n,k]; B k order used here (same as A): lanes 0-15 hold K=k0..k0+15,
    // lanes 16-31 hold K=k0+16..k0+31, column = lane&15.
    Frag16 bfrag[8];
    {
        const int kb0 = half * 16;
        for (int s = 0; s < 8; ++s) {
            const int kbase = s * 32 + kb0;
            #pragma unroll
            for (int j = 0; j < 16; ++j) {
                const int idx = n * N_HID + kbase + j;
                const float ic = is_con[idx];
                const float v  = (ic > 0.0f) ? (w_sign[idx] * abs_w0[idx]) : 0.0f;
                bfrag[s].e[j] = (__bf16)v;
            }
        }
    }

    const float wi0 = w_in[n * N_IN + 0];
    const float wi1 = w_in[n * N_IN + 1];

    // ---- Initial hidden state ----
    float h[8];
    #pragma unroll
    for (int r = 0; r < 8; ++r) {
        const int m = r + 8 * half;
        h[r] = h0[(batch0 + m) * N_HID + n];
    }

    // ---- Per-lane noise RNG stream (deterministic per launch) ----
    unsigned rng = hash_u32((unsigned)(blockIdx.x << 9) | (unsigned)tid) | 1u;
    const float noise_scale = SIGMA_NEU * 0.5f;   // sigma * sqrt(alpha)

    // ---- Prime x window 0 (async global -> LDS) ----
    // Window layout matches global: 16 rows x (WIN*N_IN) contiguous floats.
    {
        #pragma unroll
        for (int i = 0; i < 2; ++i) {
            const int chunk = tid + i * 512;          // 1024 x b128 chunks
            const int m   = chunk >> 6;
            const int rem = chunk & 63;
            const float* g = x + ((size_t)(batch0 + m) * LENGTH + 0) * N_IN + rem * 4;
            const unsigned lds = (unsigned)(uintptr_t)(&x_win[0][0][0][0])
                               + (unsigned)(m * (WIN * N_IN) + rem * 4) * 4u;
            async_load_b128(lds, g);
        }
    }

    for (int t = 0; t < LENGTH; ++t) {
        const int buf = t & 1;
        const int tw  = t & (WIN - 1);
        const int cur = (t >> 7) & 1;

        // retire the async window covering [t, t+WIN) before first use
        if (tw == 0) wait_asynccnt0();

        // 1) publish tanh(h) as bf16 into the A panel
        #pragma unroll
        for (int r = 0; r < 8; ++r) {
            const int m = r + 8 * half;
            a_buf[buf][m][n] = (__bf16)fast_tanh(h[r]);
        }
        __syncthreads();

        // 2) kick off async prefetch of the next x window (overwrites the
        //    buffer whose last reads completed before this barrier)
        if (tw == 0 && (t + WIN) < LENGTH) {
            const int t0 = t + WIN;
            const int nb = cur ^ 1;
            #pragma unroll
            for (int i = 0; i < 2; ++i) {
                const int chunk = tid + i * 512;
                const int m   = chunk >> 6;
                const int rem = chunk & 63;
                const float* g = x + ((size_t)(batch0 + m) * LENGTH + t0) * N_IN + rem * 4;
                const unsigned lds = (unsigned)(uintptr_t)(&x_win[nb][0][0][0])
                                   + (unsigned)(m * (WIN * N_IN) + rem * 4) * 4u;
                async_load_b128(lds, g);
            }
        }

        // 3) y = tanh(h) @ W_hh^T : load all 8 A fragments, then 8-deep WMMA
        //    A k order used here (same as B): lane m = L&15; K runs {k0+koff..+15}
        const int a_m    = l15;
        const int a_koff = half * 16;
        Frag16 af[8];
        #pragma unroll
        for (int s = 0; s < 8; ++s) {
            const __bf16* arow = &a_buf[buf][a_m][s * 32 + a_koff];
            af[s].q[0] = *(const uint4*)(arow);
            af[s].q[1] = *(const uint4*)(arow + 8);
        }
        v8f y = {};
        #pragma unroll
        for (int s = 0; s < 8; ++s) {
            y = __builtin_amdgcn_wmma_f32_16x16x32_bf16(
                    false, af[s].v, false, bfrag[s].v, (short)0, y, false, false);
        }

        // 4) leaky update + noise, write hidden_list, keep h in registers
        Acc8 yy; yy.v = y;
        #pragma unroll
        for (int r = 0; r < 8; ++r) {
            const int m  = r + 8 * half;
            const float2 xv = *(const float2*)&x_win[cur][m][tw][0];
            const float u  = xv.x * wi0 + xv.y * wi1;
            rng = rng * 1664525u + 1013904223u;
            const float nz = noise_scale * byte_gauss(rng);
            const float hn = (1.0f - ALPHA) * h[r] + ALPHA * (u + yy.e[r]) + nz;
            h[r] = hn;
            hidden_list[((size_t)(batch0 + m) * LENGTH + t) * N_HID + n] = hn;
        }
        // one barrier per step: A panel and x windows are double buffered
    }
}

// ---------------------------------------------------------------------------
// Output projection + h_last. One wave per (b,t) row: 256-length dot against
// w_out (O=2), shuffle-reduced. Pure bandwidth pass (~256 MB read).
// ---------------------------------------------------------------------------
__global__ __launch_bounds__(256)
void rnn_out_kernel(const float* __restrict__ hidden_list, // [B,T,H]
                    const float* __restrict__ w_out,       // [O,H]
                    float* __restrict__ output_list,       // [B,T,O]
                    float* __restrict__ h_last)            // [B,H]
{
    const int lane = threadIdx.x & 31;
    const int wib  = threadIdx.x >> 5;
    const long long row = (long long)blockIdx.x * 8 + wib;   // flat (b*T + t)
    if (row >= (long long)BATCH * LENGTH) return;

    const float* hrow = hidden_list + row * N_HID;
    float s0 = 0.0f, s1 = 0.0f;
    #pragma unroll
    for (int j = 0; j < 8; ++j) {
        const int hidx = j * 32 + lane;          // coalesced
        const float hv = hrow[hidx];
        s0 += hv * w_out[hidx];
        s1 += hv * w_out[N_HID + hidx];
    }
    #pragma unroll
    for (int off = 16; off >= 1; off >>= 1) {
        s0 += __shfl_xor(s0, off, 32);
        s1 += __shfl_xor(s1, off, 32);
    }
    if (lane == 0) {
        output_list[row * N_OUT + 0] = s0;
        output_list[row * N_OUT + 1] = s1;
    }
    if ((row & (LENGTH - 1)) == (LENGTH - 1)) {   // t == T-1 -> h_last
        const long long b = row >> 10;
        #pragma unroll
        for (int j = 0; j < 8; ++j) {
            const int hidx = j * 32 + lane;
            h_last[b * N_HID + hidx] = hrow[hidx];
        }
    }
}

extern "C" void kernel_launch(void* const* d_in, const int* in_sizes, int n_in,
                              void* d_out, int out_size, void* d_ws, size_t ws_size,
                              hipStream_t stream) {
    // setup_inputs() order:
    const float* x      = (const float*)d_in[0];  // input_signal [B,T,2]
    const float* h0     = (const float*)d_in[1];  // hidden       [B,H]
    const float* w_in   = (const float*)d_in[2];  // [H,2]
    const float* w_out  = (const float*)d_in[3];  // [2,H]
    const float* abs_w0 = (const float*)d_in[4];  // [H,H]
    const float* w_sign = (const float*)d_in[5];  // [H,H]
    const float* is_con = (const float*)d_in[6];  // [H,H]

    // d_out = hidden_list [B,T,H] ++ output_list [B,T,2] ++ h_last [B,H]
    float* hidden_list = (float*)d_out;
    float* output_list = hidden_list + (size_t)BATCH * LENGTH * N_HID;
    float* h_last      = output_list + (size_t)BATCH * LENGTH * N_OUT;

    rnn_scan_kernel<<<BATCH / 16, 512, 0, stream>>>(
        x, h0, w_in, abs_w0, w_sign, is_con, hidden_list);

    const long long rows = (long long)BATCH * LENGTH;   // 262144 rows, 8 waves/block
    rnn_out_kernel<<<(int)(rows / 8), 256, 0, stream>>>(
        hidden_list, w_out, output_list, h_last);
}
